// PDC_59674275611060
// MI455X (gfx1250) — compile-verified, matrix operand order checked by reading, not run on hardware
//
#include <hip/hip_runtime.h>

// ---------------------------------------------------------------------------
// Types for CDNA5 WMMA
// ---------------------------------------------------------------------------
typedef __attribute__((ext_vector_type(16))) __bf16 v16bf;
typedef __attribute__((ext_vector_type(8)))  float  v8f;

union FragBF {
    uint4 q[2];   // two 16-byte halves
    v16bf v;
};

#define EPSV 1e-5f
#define CIN 512
#define COUT 128
#define HH 96
#define WW 96
#define BB 8

static __device__ __forceinline__ unsigned short f32_to_bf16(float f) {
    unsigned int u = __float_as_uint(f);
    unsigned int r = (u + 0x7FFFu + ((u >> 16) & 1u)) >> 16;
    return (unsigned short)r;
}
static __device__ __forceinline__ int imin(int a, int b) { return a < b ? a : b; }
static __device__ __forceinline__ int imax(int a, int b) { return a > b ? a : b; }

// ---------------------------------------------------------------------------
// 0) Weight pre-transform: res_w[co][cin][tap] f32 -> wt2[tap][cinblk][co][ci] bf16
// ---------------------------------------------------------------------------
__global__ void wt_transform_kernel(const float* __restrict__ w,
                                    unsigned short* __restrict__ wt2) {
    int i = blockIdx.x * 256 + threadIdx.x;
    const int total = 9 * CIN * COUT;
    if (i >= total) return;
    int tap = i / (CIN * COUT);
    int r   = i % (CIN * COUT);
    int cin = r / COUT;
    int co  = r % COUT;
    float v = w[((size_t)co * CIN + cin) * 9 + tap];
    int cb = cin >> 5, ci = cin & 31;
    wt2[(((size_t)tap * 16 + cb) * COUT + co) * 32 + ci] = f32_to_bf16(v);
}

// ---------------------------------------------------------------------------
// 1) s=6 adaptive average pool (16x16 cells). block=(96), grid = B*CIN*6
// ---------------------------------------------------------------------------
__global__ void pool6_kernel(const float* __restrict__ x, float* __restrict__ p6) {
    int bid = blockIdx.x;
    int cy  = bid % 6;
    int bc  = bid / 6;           // b*CIN + c
    int t   = threadIdx.x;       // column 0..95
    const float* src = x + (size_t)bc * (HH * WW) + (cy * 16) * WW + t;
    float s = 0.f;
#pragma unroll
    for (int r = 0; r < 16; ++r) s += src[r * WW];
    __shared__ float col[WW];
    col[t] = s;
    __syncthreads();
    if (t < 6) {
        float a = 0.f;
#pragma unroll
        for (int j = 0; j < 16; ++j) a += col[t * 16 + j];
        p6[(size_t)bc * 36 + cy * 6 + t] = a * (1.0f / 256.0f);
    }
}

// ---------------------------------------------------------------------------
// 2) derive s=3,2,1 pools from the s=6 pool (exact averages of equal cells)
// ---------------------------------------------------------------------------
__global__ void pool_derive_kernel(const float* __restrict__ p6,
                                   float* __restrict__ p3,
                                   float* __restrict__ p2,
                                   float* __restrict__ p1) {
    int bc = blockIdx.x * blockDim.x + threadIdx.x;
    if (bc >= BB * CIN) return;
    float c[36];
    const float* s6 = p6 + (size_t)bc * 36;
#pragma unroll
    for (int j = 0; j < 36; ++j) c[j] = s6[j];
#pragma unroll
    for (int cy = 0; cy < 3; ++cy)
#pragma unroll
        for (int cx = 0; cx < 3; ++cx)
            p3[(size_t)bc * 9 + cy * 3 + cx] =
                0.25f * (c[(2 * cy) * 6 + 2 * cx] + c[(2 * cy) * 6 + 2 * cx + 1] +
                         c[(2 * cy + 1) * 6 + 2 * cx] + c[(2 * cy + 1) * 6 + 2 * cx + 1]);
#pragma unroll
    for (int cy = 0; cy < 2; ++cy)
#pragma unroll
        for (int cx = 0; cx < 2; ++cx) {
            float a = 0.f;
#pragma unroll
            for (int u = 0; u < 3; ++u)
#pragma unroll
                for (int w = 0; w < 3; ++w) a += c[(3 * cy + u) * 6 + 3 * cx + w];
            p2[(size_t)bc * 4 + cy * 2 + cx] = a * (1.0f / 9.0f);
        }
    float a = 0.f;
#pragma unroll
    for (int j = 0; j < 36; ++j) a += c[j];
    p1[bc] = a * (1.0f / 36.0f);
}

// ---------------------------------------------------------------------------
// 3) branch conv1 (512->128, 3x3 pad1) + BN + ReLU, tiny spatial
// ---------------------------------------------------------------------------
__global__ void branch1_kernel(const float* __restrict__ pool,
                               const float* __restrict__ w,
                               const float* __restrict__ g,
                               const float* __restrict__ bb,
                               const float* __restrict__ m,
                               const float* __restrict__ v,
                               float* __restrict__ h, int s) {
    int idx = blockIdx.x * blockDim.x + threadIdx.x;
    int total = BB * COUT * s * s;
    if (idx >= total) return;
    int xq = idx % s;
    int yq = (idx / s) % s;
    int co = (idx / (s * s)) % COUT;
    int b  = idx / (COUT * s * s);
    const float* wr = w + (size_t)co * CIN * 9;
    const float* pb = pool + (size_t)b * CIN * s * s;
    float acc = 0.f;
    for (int cin = 0; cin < CIN; ++cin) {
        const float* pc = pb + (size_t)cin * s * s;
        const float* wc = wr + (size_t)cin * 9;
#pragma unroll
        for (int tap = 0; tap < 9; ++tap) {
            int yy = yq + tap / 3 - 1;
            int xx = xq + tap % 3 - 1;
            if (yy >= 0 && yy < s && xx >= 0 && xx < s)
                acc += pc[yy * s + xx] * wc[tap];
        }
    }
    float sc = rsqrtf(v[co] + EPSV) * g[co];
    float val = (acc - m[co]) * sc + bb[co];
    h[idx] = fmaxf(val, 0.f);
}

// ---------------------------------------------------------------------------
// 4) offset conv (128->18, 3x3 pad1) + bias
// ---------------------------------------------------------------------------
__global__ void offset_kernel(const float* __restrict__ h,
                              const float* __restrict__ w,
                              const float* __restrict__ bias,
                              float* __restrict__ off, int s) {
    int idx = blockIdx.x * blockDim.x + threadIdx.x;
    int total = BB * 18 * s * s;
    if (idx >= total) return;
    int xq = idx % s;
    int yq = (idx / s) % s;
    int ch = (idx / (s * s)) % 18;
    int b  = idx / (18 * s * s);
    const float* wr = w + (size_t)ch * COUT * 9;
    const float* hb = h + (size_t)b * COUT * s * s;
    float acc = bias[ch];
    for (int cin = 0; cin < COUT; ++cin) {
        const float* hc = hb + (size_t)cin * s * s;
        const float* wc = wr + (size_t)cin * 9;
#pragma unroll
        for (int tap = 0; tap < 9; ++tap) {
            int yy = yq + tap / 3 - 1;
            int xx = xq + tap % 3 - 1;
            if (yy >= 0 && yy < s && xx >= 0 && xx < s)
                acc += hc[yy * s + xx] * wc[tap];
        }
    }
    off[idx] = acc;
}

// ---------------------------------------------------------------------------
// 5) deformable conv (DCNv1, 128->128 3x3) + BN2 + ReLU.
//    one block per (b,y,x); stage bilinear samples val[cin][tap] in LDS.
// ---------------------------------------------------------------------------
static __device__ __forceinline__ float samp(const float* hc, int s, int yy, int xx) {
    if (yy < 0 || yy >= s || xx < 0 || xx >= s) return 0.f;
    return hc[yy * s + xx];
}

__global__ __launch_bounds__(128) void deform_kernel(
    const float* __restrict__ h, const float* __restrict__ off,
    const float* __restrict__ w, const float* __restrict__ bias,
    const float* __restrict__ g, const float* __restrict__ bb,
    const float* __restrict__ m, const float* __restrict__ v,
    float* __restrict__ h2, int s) {
    __shared__ float val[COUT * 9];
    int b   = blockIdx.x / (s * s);
    int pos = blockIdx.x % (s * s);
    int y = pos / s, xq = pos % s;
    int t = threadIdx.x;  // cin in stage1, co in stage2
    const float* hc = h + ((size_t)b * COUT + t) * s * s;
    const float* ob = off + (size_t)b * 18 * s * s;
#pragma unroll
    for (int tap = 0; tap < 9; ++tap) {
        float oy = ob[(2 * tap) * s * s + pos];
        float ox = ob[(2 * tap + 1) * s * s + pos];
        float py = (float)(y + tap / 3 - 1) + oy;
        float px = (float)(xq + tap % 3 - 1) + ox;
        float fy = floorf(py), fx = floorf(px);
        int y0 = (int)fy, x0 = (int)fx;
        float wy = py - fy, wx = px - fx;
        float v00 = samp(hc, s, y0, x0);
        float v01 = samp(hc, s, y0, x0 + 1);
        float v10 = samp(hc, s, y0 + 1, x0);
        float v11 = samp(hc, s, y0 + 1, x0 + 1);
        val[t * 9 + tap] =
            (v00 * (1.f - wy) + v10 * wy) * (1.f - wx) +
            (v01 * (1.f - wy) + v11 * wy) * wx;
    }
    __syncthreads();
    float acc = 0.f;
    const float* wr = w + (size_t)t * COUT * 9;  // [co][cin*9+tap]
    for (int j = 0; j < COUT * 9; ++j) acc += val[j] * wr[j];
    acc += bias[t];
    float sc = rsqrtf(v[t] + EPSV) * g[t];
    float r = (acc - m[t]) * sc + bb[t];
    h2[((size_t)b * COUT + t) * s * s + pos] = fmaxf(r, 0.f);
}

// ---------------------------------------------------------------------------
// 6) Main 3x3 conv 512->128 via bf16 WMMA implicit GEMM, fused BN+ReLU and
//    fused bilinear upsample-add of the 4 branch outputs.
//    block = 256 thr (8 waves), tile = [128 cout x 32 px], grid = B*H*(W/32).
//    Per-tap LDS staging: [32 px][512 cin] bf16 = 32KB -> only 9 barrier
//    pairs per block; inner cin-block loop is barrier-free WMMA issue.
// ---------------------------------------------------------------------------
__global__ __launch_bounds__(256) void main_conv_wmma_kernel(
    const float* __restrict__ x, const unsigned short* __restrict__ wt2,
    const float* __restrict__ g, const float* __restrict__ bbias,
    const float* __restrict__ m, const float* __restrict__ v,
    const float* __restrict__ h2_1, const float* __restrict__ h2_2,
    const float* __restrict__ h2_3, const float* __restrict__ h2_6,
    float* __restrict__ out) {
    __shared__ __align__(16) unsigned short bsh[32 * CIN];  // [pixel][cin] bf16, 32KB
    __shared__ float s_scale[COUT];
    __shared__ float s_bias[COUT];

    int tid = threadIdx.x;
    int bid = blockIdx.x;
    int b  = bid / (HH * 3);
    int r  = bid % (HH * 3);
    int y  = r / 3;
    int x0 = (r % 3) * 32;

    if (tid < COUT) {
        float sc = rsqrtf(v[tid] + EPSV) * g[tid];
        s_scale[tid] = sc;
        s_bias[tid]  = bbias[tid] - m[tid] * sc;
    }

    int wv   = tid >> 5;      // wave 0..7  -> cout strip
    int lane = tid & 31;
    int mloc = lane & 15;
    int hh   = lane >> 4;     // lane half selects k-half per WMMA 16-bit layout

    v8f acc0 = {};
    v8f acc1 = {};

    // loader mapping: pixel = tid&31, cin group of 64 = tid>>5
    int lp = tid & 31;
    int cg = tid >> 5;
    const float* xb = x + (size_t)b * CIN * (HH * WW);

    for (int tap = 0; tap < 9; ++tap) {
        int ky = tap / 3 - 1, kx = tap % 3 - 1;
        int yy = y + ky;
        int xx = x0 + kx + lp;
        bool ok = (yy >= 0) && (yy < HH) && (xx >= 0) && (xx < WW);
        const float* src = xb + (size_t)yy * WW + xx;

        __syncthreads();   // previous tap's tile fully consumed
        // --- fill activation tile (f32 -> bf16), pixel-major, 8 x b128 ---
        for (int jj = 0; jj < 8; ++jj) {
            int cbase = cg * 64 + jj * 8;
            if (jj < 7)
                __builtin_prefetch(src + (size_t)(cbase + 8) * (HH * WW), 0, 0);
            unsigned int pk[4];
#pragma unroll
            for (int u = 0; u < 4; ++u) {
                float fa = ok ? src[(size_t)(cbase + 2 * u) * (HH * WW)] : 0.f;
                float fb = ok ? src[(size_t)(cbase + 2 * u + 1) * (HH * WW)] : 0.f;
                pk[u] = (unsigned int)f32_to_bf16(fa) |
                        ((unsigned int)f32_to_bf16(fb) << 16);
            }
            uint4 q;
            q.x = pk[0]; q.y = pk[1]; q.z = pk[2]; q.w = pk[3];
            *(uint4*)&bsh[lp * CIN + cbase] = q;
        }
        __syncthreads();

        // --- barrier-free inner loop over 16 cin blocks ---
        for (int cb = 0; cb < 16; ++cb) {
            // A fragment (weights) straight from L2-resident wt2
            const unsigned short* W0 =
                wt2 + (((size_t)tap * 16 + cb) * COUT + wv * 16 + mloc) * 32;
            FragBF a;
            a.q[0] = *(const uint4*)(W0 + hh * 8);
            a.q[1] = *(const uint4*)(W0 + 16 + hh * 8);

            // B fragments from LDS, two pixel tiles
            int kb = cb * 32 + hh * 8;
            FragBF b0, b1;
            b0.q[0] = *(const uint4*)&bsh[mloc * CIN + kb];
            b0.q[1] = *(const uint4*)&bsh[mloc * CIN + kb + 16];
            b1.q[0] = *(const uint4*)&bsh[(16 + mloc) * CIN + kb];
            b1.q[1] = *(const uint4*)&bsh[(16 + mloc) * CIN + kb + 16];

            acc0 = __builtin_amdgcn_wmma_f32_16x16x32_bf16(
                false, a.v, false, b0.v, (short)0, acc0, false, false);
            acc1 = __builtin_amdgcn_wmma_f32_16x16x32_bf16(
                false, a.v, false, b1.v, (short)0, acc1, false, false);
        }
    }

    // ---------------- epilogue: BN + ReLU + 4x bilinear upsample-add --------
    const float* h2p[4] = {h2_1, h2_2, h2_3, h2_6};
    const int ssv[4]    = {1, 2, 3, 6};
    int sy0[4], sy1[4];
    float wyv[4];
#pragma unroll
    for (int i = 0; i < 4; ++i) {
        int s = ssv[i];
        float srcy = ((float)y + 0.5f) * (float)s * (1.0f / 96.0f) - 0.5f;
        float fl = floorf(srcy);
        int i0 = (int)fl;
        wyv[i] = srcy - fl;
        sy0[i] = imin(imax(i0, 0), s - 1);
        sy1[i] = imin(imax(i0 + 1, 0), s - 1);
    }

#pragma unroll
    for (int t = 0; t < 2; ++t) {
        v8f acc = t ? acc1 : acc0;
        int px = x0 + t * 16 + mloc;
        int sx0[4], sx1[4];
        float wxv[4];
#pragma unroll
        for (int i = 0; i < 4; ++i) {
            int s = ssv[i];
            float srcx = ((float)px + 0.5f) * (float)s * (1.0f / 96.0f) - 0.5f;
            float fl = floorf(srcx);
            int j0 = (int)fl;
            wxv[i] = srcx - fl;
            sx0[i] = imin(imax(j0, 0), s - 1);
            sx1[i] = imin(imax(j0 + 1, 0), s - 1);
        }
#pragma unroll
        for (int rr = 0; rr < 8; ++rr) {
            int co = wv * 16 + rr + 8 * hh;
            float val = acc[rr] * s_scale[co] + s_bias[co];
            val = fmaxf(val, 0.f);
#pragma unroll
            for (int i = 0; i < 4; ++i) {
                int s = ssv[i];
                const float* hr0 =
                    h2p[i] + ((size_t)(b * COUT + co) * s + sy0[i]) * s;
                const float* hr1 =
                    h2p[i] + ((size_t)(b * COUT + co) * s + sy1[i]) * s;
                float v00 = hr0[sx0[i]], v01 = hr0[sx1[i]];
                float v10 = hr1[sx0[i]], v11 = hr1[sx1[i]];
                float wy = wyv[i], wx = wxv[i];
                val += (v00 * (1.f - wy) + v10 * wy) * (1.f - wx) +
                       (v01 * (1.f - wy) + v11 * wy) * wx;
            }
            out[((size_t)(b * COUT + co) * HH + y) * WW + px] = val;
        }
    }
}

// ---------------------------------------------------------------------------
// Host launcher
// ---------------------------------------------------------------------------
extern "C" void kernel_launch(void* const* d_in, const int* in_sizes, int n_in,
                              void* d_out, int out_size, void* d_ws, size_t ws_size,
                              hipStream_t stream) {
    (void)in_sizes; (void)n_in; (void)out_size; (void)ws_size;

    const float* x     = (const float*)d_in[0];
    const float* res_w = (const float*)d_in[1];
    const float* res_g = (const float*)d_in[2];
    const float* res_b = (const float*)d_in[3];
    const float* res_m = (const float*)d_in[4];
    const float* res_v = (const float*)d_in[5];
    const float* br_w  = (const float*)d_in[6];
    const float* br1_g = (const float*)d_in[7];
    const float* br1_b = (const float*)d_in[8];
    const float* br1_m = (const float*)d_in[9];
    const float* br1_v = (const float*)d_in[10];
    const float* off_w = (const float*)d_in[11];
    const float* off_b = (const float*)d_in[12];
    const float* dcn_w = (const float*)d_in[13];
    const float* dcn_b = (const float*)d_in[14];
    const float* br2_g = (const float*)d_in[15];
    const float* br2_b = (const float*)d_in[16];
    const float* br2_m = (const float*)d_in[17];
    const float* br2_v = (const float*)d_in[18];
    float* out = (float*)d_out;

    // ---- workspace layout ----
    char* w8 = (char*)d_ws;
    unsigned short* wt2 = (unsigned short*)w8;          // 9*512*128 bf16
    float* p6 = (float*)(w8 + (size_t)9 * 512 * 128 * 2);
    float* p3 = p6 + (size_t)8 * 512 * 36;
    float* p2 = p3 + (size_t)8 * 512 * 9;
    float* p1 = p2 + (size_t)8 * 512 * 4;
    float* hq = p1 + (size_t)8 * 512;

    const int ss[4] = {1, 2, 3, 6};
    const float* pools[4] = {p1, p2, p3, p6};
    float* h_buf[4];
    float* o_buf[4];
    float* g_buf[4];
    size_t off_f = 0;
    for (int i = 0; i < 4; ++i) { h_buf[i] = hq + off_f; off_f += (size_t)8 * 128 * ss[i] * ss[i]; }
    for (int i = 0; i < 4; ++i) { o_buf[i] = hq + off_f; off_f += (size_t)8 * 18  * ss[i] * ss[i]; }
    for (int i = 0; i < 4; ++i) { g_buf[i] = hq + off_f; off_f += (size_t)8 * 128 * ss[i] * ss[i]; }

    // 0) weight transform for the WMMA main conv
    {
        int total = 9 * 512 * 128;
        wt_transform_kernel<<<(total + 255) / 256, 256, 0, stream>>>(res_w, wt2);
    }
    // 1) s=6 pool (single pass over x), 2) derive s=3/2/1
    pool6_kernel<<<8 * 512 * 6, 96, 0, stream>>>(x, p6);
    pool_derive_kernel<<<(8 * 512 + 255) / 256, 256, 0, stream>>>(p6, p3, p2, p1);

    // 3-5) per-setting branch pipeline (tiny)
    for (int i = 0; i < 4; ++i) {
        int s = ss[i];
        int t1 = 8 * 128 * s * s;
        branch1_kernel<<<(t1 + 127) / 128, 128, 0, stream>>>(
            pools[i], br_w + (size_t)i * 128 * 512 * 9,
            br1_g + i * 128, br1_b + i * 128, br1_m + i * 128, br1_v + i * 128,
            h_buf[i], s);
        int t2 = 8 * 18 * s * s;
        offset_kernel<<<(t2 + 127) / 128, 128, 0, stream>>>(
            h_buf[i], off_w + (size_t)i * 18 * 128 * 9, off_b + i * 18,
            o_buf[i], s);
        deform_kernel<<<8 * s * s, 128, 0, stream>>>(
            h_buf[i], o_buf[i], dcn_w + (size_t)i * 128 * 128 * 9,
            dcn_b + i * 128, br2_g + i * 128, br2_b + i * 128,
            br2_m + i * 128, br2_v + i * 128, g_buf[i], s);
    }

    // 6) main WMMA conv + fused BN/ReLU + fused 4x bilinear upsample-add
    main_conv_wmma_kernel<<<8 * 96 * 3, 256, 0, stream>>>(
        x, wt2, res_g, res_b, res_m, res_v,
        g_buf[0], g_buf[1], g_buf[2], g_buf[3], out);
}
